// LinearVAECell_61366492725365
// MI455X (gfx1250) — matrix operand order repaired and checked by reading, not compiled, not run on hardware
//
#include <hip/hip_runtime.h>
#include <math.h>

// ---------------------------------------------------------------------------
// Problem constants (from the reference)
// ---------------------------------------------------------------------------
#define Bq      32
#define Tfull   40
#define Tm1     39
#define Vv      30000
#define ESz     300
#define EMBz    300
#define NST     10
#define Hh      210
#define NUq     9

#define KX      1120          // padded xin K for decoder GEMM (900 + 210 -> 1120)
#define NG      848           // padded gate N (840 -> 848 = 53 tiles)
#define NT_DEC  53
#define KPAD_H  224           // 210 -> 224 (7 k-chunks of 32)
#define KPAD_B  416           // 400 -> 416 (13 k-chunks)

typedef _Float16 v8h  __attribute__((ext_vector_type(8)));
typedef _Float16 v16h __attribute__((ext_vector_type(16)));
typedef float    v8f  __attribute__((ext_vector_type(8)));

__device__ __forceinline__ v8f zero8() {
  v8f z = {0.f,0.f,0.f,0.f,0.f,0.f,0.f,0.f};
  return z;
}
__device__ __forceinline__ v16h frag16(const _Float16* p0, const _Float16* p1) {
  v8h a = *(const v8h*)p0;
  v8h b = *(const v8h*)p1;
  return __builtin_shufflevector(a, b, 0,1,2,3,4,5,6,7,8,9,10,11,12,13,14,15);
}
__device__ __forceinline__ v8f wmma_f16(v16h a, v16h b, v8f c) {
  // D = A(16x32,f16) x B(32x16,f16) + C(16x16,f32)
  return __builtin_amdgcn_wmma_f32_16x16x32_f16(false, a, false, b, (short)0, c, false, false);
}
__device__ __forceinline__ float sigf(float x) { return 1.0f / (1.0f + expf(-x)); }
__device__ __forceinline__ float lse2(float x, float y) {
  float m = fmaxf(x, y);
  return m + logf(expf(x - m) + expf(y - m));
}

// ---------------------------------------------------------------------------
// Generic WMMA GEMM:  C[M,N] = A[M,Kp](f16) * Bt[N,Kp](f16)^T + bias[N]
// A row-major padded-K; Bt is W transposed so a B-fragment lane reads two
// runs of 8 contiguous K of its column, at k0+8h and k0+16+8h. 256 thr = 8 waves; wave = 16x64 strip.
// Boundary N-tiles are CLAMPED for loads (always in-bounds) so the K-loop has
// no EXEC manipulation around the WMMAs; only the stores are guarded.
// ---------------------------------------------------------------------------
__global__ __launch_bounds__(256) void k_wmma_gemm_bias(
    const _Float16* __restrict__ A, const _Float16* __restrict__ Bt,
    const float* __restrict__ bias, float* __restrict__ C,
    int M, int N, int Kp)
{
  const int lane = threadIdx.x & 31;
  const int wid  = threadIdx.x >> 5;
  const int l15  = lane & 15;
  const int hi   = lane >> 4;
  const int Ntiles = N >> 4;
  const int strip  = blockIdx.x * 8 + wid;
  const int mt     = blockIdx.y;
  const int nt0    = strip * 4;
  if (nt0 >= Ntiles) return;

  v8f acc[4];
  float bv[4];
  const _Float16* brow[4];
#pragma unroll
  for (int q = 0; q < 4; ++q) {
    acc[q] = zero8();
    int nt = nt0 + q;
    int ntc = (nt < Ntiles) ? nt : (Ntiles - 1);      // clamp: loads stay valid
    bv[q]   = bias[ntc * 16 + l15];
    brow[q] = Bt + (size_t)(ntc * 16 + l15) * Kp;
  }
  const _Float16* arow = A + (size_t)(mt * 16 + l15) * Kp;
  const int kboff = hi << 3;
  const int kaoff = hi << 3;
#pragma unroll 2
  for (int k0 = 0; k0 < Kp; k0 += 32) {
    const _Float16* ap = arow + k0 + kaoff;
    v16h af = frag16(ap, ap + 16);
#pragma unroll
    for (int q = 0; q < 4; ++q) {
      const _Float16* bp = brow[q] + k0 + kboff;
      v16h bf = frag16(bp, bp + 16);
      acc[q] = wmma_f16(af, bf, acc[q]);
    }
  }
#pragma unroll
  for (int q = 0; q < 4; ++q) {
    int nt = nt0 + q;
    if (nt >= Ntiles) continue;                        // store-only guard
    const int n = nt * 16 + l15;
#pragma unroll
    for (int e = 0; e < 8; ++e) {
      int mr = mt * 16 + e + (hi << 3);
      C[(size_t)mr * N + n] = acc[q][e] + bv[q];
    }
  }
}

// ---------------------------------------------------------------------------
// Small helper kernels (tiny layers; VALU is the right tool here)
// ---------------------------------------------------------------------------
__global__ void k_sgemm_act(const float* __restrict__ A, const float* __restrict__ W,
                            const float* __restrict__ bias, float* __restrict__ C,
                            int M, int K, int N, int act)
{
  int idx = blockIdx.x * blockDim.x + threadIdx.x;
  if (idx >= M * N) return;
  int m = idx / N, n = idx % N;
  const float* a = A + (size_t)m * K;
  float s = bias[n];
  for (int k = 0; k < K; ++k) s += a[k] * W[(size_t)k * N + n];
  if (act == 1) s = fmaxf(s, 0.0f);
  else if (act == 2) s = tanhf(s);
  C[idx] = s;
}

__global__ void k_concat2(const float* __restrict__ A, int K1,
                          const float* __restrict__ Bp, int K2,
                          float* __restrict__ Out, int M)
{
  int K = K1 + K2;
  int idx = blockIdx.x * blockDim.x + threadIdx.x;
  if (idx >= M * K) return;
  int m = idx / K, k = idx % K;
  Out[idx] = (k < K1) ? A[m * K1 + k] : Bp[m * K2 + (k - K1)];
}

__global__ void k_enc_softmax(const float* __restrict__ logits, const float* __restrict__ gum,
                              const float* __restrict__ tau,
                              float* __restrict__ z_out, float* __restrict__ q_out,
                              float* __restrict__ lq_out, float* __restrict__ z_ws)
{
  int b = blockIdx.x * blockDim.x + threadIdx.x;
  if (b >= Bq) return;
  float l[NST];
  float mx = -1e30f;
  for (int j = 0; j < NST; ++j) { l[j] = logits[b * NST + j]; mx = fmaxf(mx, l[j]); }
  float se = 0.f;
  for (int j = 0; j < NST; ++j) se += expf(l[j] - mx);
  float lse = mx + logf(se);
  for (int j = 0; j < NST; ++j) {
    q_out[b * NST + j]  = expf(l[j] - lse);
    lq_out[b * NST + j] = l[j] - lse;
  }
  float t = tau[0];
  float g2[NST];
  float mx2 = -1e30f;
  for (int j = 0; j < NST; ++j) { g2[j] = (l[j] + gum[b * NST + j]) / t; mx2 = fmaxf(mx2, g2[j]); }
  float se2 = 0.f;
  for (int j = 0; j < NST; ++j) se2 += expf(g2[j] - mx2);
  for (int j = 0; j < NST; ++j) {
    float z = expf(g2[j] - mx2) / se2;
    z_out[b * NST + j] = z;
    z_ws[b * NST + j]  = z;
  }
}

__global__ void k_pz(const float* __restrict__ trlog, float* __restrict__ pz,
                     float* __restrict__ logpz)
{
  int b = blockIdx.x * blockDim.x + threadIdx.x;
  if (b >= Bq) return;
  float l[NST];
  float mx = -1e30f;
  for (int j = 0; j < NST; ++j) { l[j] = trlog[b * NST + j]; mx = fmaxf(mx, l[j]); }
  float se = 0.f;
  for (int j = 0; j < NST; ++j) se += expf(l[j] - mx);
  for (int j = 0; j < NST; ++j) {
    float p = expf(l[j] - mx) / se;
    pz[b * NST + j] = p;
    logpz[b * NST + j] = logf(p + 1e-20f);
  }
}

__global__ void k_state_lstm(const float* __restrict__ net2, const float* __restrict__ inputs,
                             const float* __restrict__ h0, const float* __restrict__ c0,
                             const float* __restrict__ wih, const float* __restrict__ whh,
                             const float* __restrict__ bih, const float* __restrict__ bhh,
                             float* __restrict__ nh, float* __restrict__ nc)
{
  int idx = blockIdx.x * blockDim.x + threadIdx.x;
  if (idx >= Bq * NST) return;
  int b = idx / NST, j = idx % NST;
  float gi = 0.f, gf = 0.f, gg = 0.f, go = 0.f;
  for (int k = 0; k < 800; ++k) {
    float x = (k < 200) ? net2[b * 200 + k] : inputs[b * 600 + (k - 200)];
    gi += x * wih[(0 * NST + j) * 800 + k];
    gf += x * wih[(1 * NST + j) * 800 + k];
    gg += x * wih[(2 * NST + j) * 800 + k];
    go += x * wih[(3 * NST + j) * 800 + k];
  }
  for (int kk = 0; kk < NST; ++kk) {
    float h = h0[b * NST + kk];
    gi += h * whh[(0 * NST + j) * NST + kk];
    gf += h * whh[(1 * NST + j) * NST + kk];
    gg += h * whh[(2 * NST + j) * NST + kk];
    go += h * whh[(3 * NST + j) * NST + kk];
  }
  gi += bih[j]            + bhh[j];
  gf += bih[NST + j]      + bhh[NST + j];
  gg += bih[2 * NST + j]  + bhh[2 * NST + j];
  go += bih[3 * NST + j]  + bhh[3 * NST + j];
  float c = sigf(gf) * c0[idx] + sigf(gi) * tanhf(gg);
  float h = sigf(go) * tanhf(c);
  nh[idx] = h;
  nc[idx] = c;
}

// ---------------------------------------------------------------------------
// Prep kernels for the decoder + big GEMMs
// ---------------------------------------------------------------------------
__global__ void k_xsum(const float* __restrict__ q, float* __restrict__ Xsum)
{
  int idx = blockIdx.x * blockDim.x + threadIdx.x;
  if (idx >= Bq * NUq * 2 * Hh) return;
  int c = idx % Hh;
  int r = idx / Hh;
  int i = r % 2; r /= 2;
  int n = r % NUq;
  int b = r / NUq;
  // X_prev = q[:, :9], X_cur = q[:, 1:10]
  float vp = q[(((size_t)b * (NUq + 1) + n)     * 2 + i) * Hh + c];
  float vc = q[(((size_t)b * (NUq + 1) + n + 1) * 2 + i) * Hh + c];
  Xsum[idx] = vp + vc;
}

__global__ void k_xpxc(const float* __restrict__ q, float* __restrict__ XpXc)
{
  int idx = blockIdx.x * blockDim.x + threadIdx.x;
  if (idx >= Bq * NUq * 4) return;
  int j = idx % 2;
  int t = idx / 2;
  int i = t % 2; t /= 2;
  int n = t % NUq;
  int b = t / NUq;
  const float* xp = q + (((size_t)b * (NUq + 1) + n)     * 2 + i) * Hh;
  const float* xc = q + (((size_t)b * (NUq + 1) + n + 1) * 2 + j) * Hh;
  float s = 0.f;
  for (int c = 0; c < Hh; ++c) s += xp[c] * xc[c];
  XpXc[idx] = s;
}

// Wcat[j,k] f16: [848,1120]: k<900 -> wih[j,k]; 900<=k<1110 -> whh[j,k-900]; else 0
__global__ void k_wcat(const float* __restrict__ wih, const float* __restrict__ whh,
                       _Float16* __restrict__ Wcat)
{
  int idx = blockIdx.x * blockDim.x + threadIdx.x;
  if (idx >= NG * KX) return;
  int j = idx / KX, k = idx % KX;
  float v = 0.f;
  if (j < 4 * Hh) {
    if (k < 900)       v = wih[(size_t)j * 900 + k];
    else if (k < 1110) v = whh[(size_t)j * Hh + (k - 900)];
  }
  Wcat[idx] = (_Float16)v;
}

// Bt[n,k] = (k<K) ? W[k,n] : 0   (f32 [K,N] -> f16 [N,Kp])
__global__ void k_transpose_f16(const float* __restrict__ W, _Float16* __restrict__ Bt,
                                int K, int N, int Kp)
{
  size_t idx = (size_t)blockIdx.x * blockDim.x + threadIdx.x;
  if (idx >= (size_t)N * Kp) return;
  int n = (int)(idx / Kp);
  int k = (int)(idx % Kp);
  Bt[idx] = (k < K) ? (_Float16)W[(size_t)k * N + n] : (_Float16)0.0f;
}

__global__ void k_rows_to_f16(const float* __restrict__ X, _Float16* __restrict__ Y,
                              int M, int K, int Kp)
{
  int idx = blockIdx.x * blockDim.x + threadIdx.x;
  if (idx >= M * Kp) return;
  int r = idx / Kp, k = idx % Kp;
  Y[idx] = (k < K) ? (_Float16)X[(size_t)r * K + k] : (_Float16)0.0f;
}

__global__ void k_zero_f16(_Float16* __restrict__ p, int n)
{
  int i = blockIdx.x * blockDim.x + threadIdx.x;
  if (i < n) p[i] = (_Float16)0.0f;
}

// ---------------------------------------------------------------------------
// Persistent decoder: both 39-step LSTM+CRF decoders, one 1024-thread block.
// LDS layout (bytes): xin f16[32][1120] @0 (71680); g f32[32][848] @71680
// (108544); h f32[32][210] @180224; c @207104; s[32][18] @233984; m[32][9]
// @236288; total 237440 < 320KB/WGP.
// ---------------------------------------------------------------------------
#define DEC_LDS_BYTES 237440

__global__ __launch_bounds__(1024) void k_decoder(
    const float* __restrict__ dec_in1,   // [32,210] (h0 and dec1 c0)
    const float* __restrict__ emb1,      // dec_emb1 [32,40,300]
    const float* __restrict__ emb2,
    const float* __restrict__ prev_emb,  // [32,9,600]
    const float* __restrict__ Xsum,      // [32,9,2,210]
    const float* __restrict__ XpXc,      // [32,9,2,2]
    const _Float16* __restrict__ Wcat1, const _Float16* __restrict__ Wcat2,
    const float* __restrict__ bih1, const float* __restrict__ bhh1,
    const float* __restrict__ bih2, const float* __restrict__ bhh2,
    _Float16* __restrict__ A1,           // [1248,224] f16, pre-zeroed
    _Float16* __restrict__ A2,
    float* __restrict__ h1_out)          // [32,210]
{
  extern __shared__ char smem[];
  _Float16* xinL = (_Float16*)(smem + 0);
  float*    gL   = (float*)(smem + 71680);
  float*    hL   = (float*)(smem + 180224);
  float*    cL   = (float*)(smem + 207104);
  float*    sL   = (float*)(smem + 233984);
  float*    mL   = (float*)(smem + 236288);

  const int tid  = threadIdx.x;
  const int lane = tid & 31;
  const int wid  = tid >> 5;
  const int l15  = lane & 15;
  const int hi   = lane >> 4;

  for (int dec = 0; dec < 2; ++dec) {
    const _Float16* Wc  = dec ? Wcat2 : Wcat1;
    const float*    bi  = dec ? bih2 : bih1;
    const float*    bh  = dec ? bhh2 : bhh1;
    const float*    emb = dec ? emb2 : emb1;
    _Float16*       Ad  = dec ? A2 : A1;

    for (int i = tid; i < Bq * Hh; i += 1024) {
      hL[i] = dec_in1[i];
      if (dec == 0) cL[i] = dec_in1[i];   // dec2 keeps c1 from dec1
    }
    __syncthreads();

    for (int t = 0; t < Tm1; ++t) {
      // --- s[b,n,i] = Xsum[b,n,i,:] . h[b,:] ---
      for (int i = tid; i < Bq * 18; i += 1024) {
        int b = i / 18;
        const float* xs = Xsum + (size_t)i * Hh;
        const float* hb = hL + b * Hh;
        float a = 0.f;
        for (int c = 0; c < Hh; ++c) a += xs[c] * hb[c];
        sL[i] = a;
      }
      __syncthreads();

      // --- CRF node marginals (C=2, N=9), one thread per b ---
      if (tid < Bq) {
        int b = tid;
        const float* xp = XpXc + b * 36;
        const float* sb = sL + b * 18;
#define PHI(n,i2,j2) (xp[(n)*4 + (i2)*2 + (j2)] + sb[(n)*2 + (i2)])
        float al0[NUq], al1[NUq], be1[NUq];
        float a0 = 0.f, a1 = 0.f;
#pragma unroll
        for (int n = 0; n < NUq; ++n) {
          al0[n] = a0; al1[n] = a1;
          float n0 = lse2(a0 + PHI(n,0,0), a1 + PHI(n,0,1));
          float n1 = lse2(a0 + PHI(n,1,0), a1 + PHI(n,1,1));
          a0 = n0; a1 = n1;
        }
        float logZ = lse2(a0, a1);
        float b0 = 0.f, b1 = 0.f;
#pragma unroll
        for (int n = NUq - 1; n >= 0; --n) {
          be1[n] = b1;                     // beta[n][i=1]
          float n0 = lse2(b0 + PHI(n,0,0), b1 + PHI(n,1,0));
          float n1 = lse2(b0 + PHI(n,0,1), b1 + PHI(n,1,1));
          b0 = n0; b1 = n1;
        }
#pragma unroll
        for (int n = 0; n < NUq; ++n) {
          float e0 = expf(al0[n] + PHI(n,1,0) + be1[n] - logZ);
          float e1 = expf(al1[n] + PHI(n,1,1) + be1[n] - logZ);
          mL[b * NUq + n] = e0 + e1;
        }
#undef PHI
      }
      __syncthreads();

      // --- build xin f16 [32,1120] = [x_t | context | h | pad] ---
      for (int i = tid; i < Bq * KX; i += 1024) {
        int b = i / KX, k = i % KX;
        float v;
        if (k < EMBz) {
          v = emb[((size_t)b * Tfull + t) * EMBz + k];
        } else if (k < 900) {
          int d = k - EMBz;
          const float* pe = prev_emb + (size_t)b * NUq * (2 * ESz) + d;
          const float* mb = mL + b * NUq;
          float a = 0.f;
#pragma unroll
          for (int n = 0; n < NUq; ++n) a += mb[n] * pe[(size_t)n * (2 * ESz)];
          v = a * (1.0f / (float)NUq);
        } else if (k < 1110) {
          v = hL[b * Hh + (k - 900)];
        } else {
          v = 0.f;
        }
        xinL[i] = (_Float16)v;
      }
      __syncthreads();

      // --- WMMA GEMM: g[32,840] = xin[32,1120] @ Wcat^T ---
      for (int task = wid; task < 2 * NT_DEC; task += 32) {
        int mt = task / NT_DEC, nt = task % NT_DEC;
        v8f acc = zero8();
        const _Float16* arow = xinL + (size_t)(mt * 16 + l15) * KX;
        const _Float16* brow = Wc + (size_t)(nt * 16 + l15) * KX;
#pragma unroll 2
        for (int k0 = 0; k0 < KX; k0 += 32) {
          v16h af = frag16(arow + k0 + (hi << 3), arow + k0 + (hi << 3) + 16);
          v16h bf = frag16(brow + k0 + (hi << 3), brow + k0 + (hi << 3) + 16);
          acc = wmma_f16(af, bf, acc);
        }
        int n = nt * 16 + l15;
#pragma unroll
        for (int e = 0; e < 8; ++e)
          gL[(mt * 16 + e + (hi << 3)) * NG + n] = acc[e];
      }
      __syncthreads();

      // --- gates, state update, emit h as padded f16 GEMM row ---
      for (int i = tid; i < Bq * Hh; i += 1024) {
        int b = i / Hh, j = i % Hh;
        const float* gb = gL + b * NG;
        float iv = gb[j]           + bi[j]           + bh[j];
        float fv = gb[Hh + j]      + bi[Hh + j]      + bh[Hh + j];
        float gv = gb[2 * Hh + j]  + bi[2 * Hh + j]  + bh[2 * Hh + j];
        float ov = gb[3 * Hh + j]  + bi[3 * Hh + j]  + bh[3 * Hh + j];
        float cn = sigf(fv) * cL[i] + sigf(iv) * tanhf(gv);
        float hn = sigf(ov) * tanhf(cn);
        cL[i] = cn;
        hL[i] = hn;
        Ad[(size_t)(b * Tm1 + t) * KPAD_H + j] = (_Float16)hn;
      }
      __syncthreads();
    }

    if (dec == 0) {
      for (int i = tid; i < Bq * Hh; i += 1024) h1_out[i] = hL[i];
      __syncthreads();
    }
  }
}

// ---------------------------------------------------------------------------
// Host: orchestrate the whole forward pass on `stream`
// ---------------------------------------------------------------------------
static inline unsigned cdiv(size_t a, unsigned b) { return (unsigned)((a + b - 1) / b); }

extern "C" void kernel_launch(void* const* d_in, const int* in_sizes, int n_in,
                              void* d_out, int out_size, void* d_ws, size_t ws_size,
                              hipStream_t stream)
{
  (void)in_sizes; (void)n_in; (void)out_size; (void)ws_size;

  const float* inputs  = (const float*)d_in[0];   // [32,600]
  const float* state_h = (const float*)d_in[1];   // [32,10]
  const float* state_c = (const float*)d_in[2];   // [32,10]
  const float* emb1    = (const float*)d_in[3];   // [32,40,300]
  const float* emb2    = (const float*)d_in[4];
  const float* prevemb = (const float*)d_in[5];   // [32,9,600]
  const float* query   = (const float*)d_in[6];   // [32,10,2,210]
  const float* gumbel  = (const float*)d_in[7];   // [32,10]
  const int P = 8;                                // params (insertion order)
  const float* p_tau      = (const float*)d_in[P + 0];
  const float* enc_w1     = (const float*)d_in[P + 1];
  const float* enc_b1     = (const float*)d_in[P + 2];
  const float* enc_w2     = (const float*)d_in[P + 3];
  const float* enc_b2     = (const float*)d_in[P + 4];
  const float* enc_fc_w   = (const float*)d_in[P + 5];
  const float* enc_fc_b   = (const float*)d_in[P + 6];
  const float* dec_w1     = (const float*)d_in[P + 7];
  const float* dec_b1     = (const float*)d_in[P + 8];
  const float* dec_w2     = (const float*)d_in[P + 9];
  const float* dec_b2     = (const float*)d_in[P + 10];
  const float* d1_wih     = (const float*)d_in[P + 11];
  const float* d1_whh     = (const float*)d_in[P + 12];
  const float* d1_bih     = (const float*)d_in[P + 13];
  const float* d1_bhh     = (const float*)d_in[P + 14];
  const float* d2_wih     = (const float*)d_in[P + 15];
  const float* d2_whh     = (const float*)d_in[P + 16];
  const float* d2_bih     = (const float*)d_in[P + 17];
  const float* d2_bhh     = (const float*)d_in[P + 18];
  const float* fc1_w      = (const float*)d_in[P + 19];
  const float* fc1_b      = (const float*)d_in[P + 20];
  const float* fc2_w      = (const float*)d_in[P + 21];
  const float* fc2_b      = (const float*)d_in[P + 22];
  const float* bow_fc1_w  = (const float*)d_in[P + 23];
  const float* bow_fc1_b  = (const float*)d_in[P + 24];
  const float* bow_p1_w   = (const float*)d_in[P + 25];
  const float* bow_p1_b   = (const float*)d_in[P + 26];
  const float* bow_fc2_w  = (const float*)d_in[P + 27];
  const float* bow_fc2_b  = (const float*)d_in[P + 28];
  const float* bow_p2_w   = (const float*)d_in[P + 29];
  const float* bow_p2_b   = (const float*)d_in[P + 30];
  const float* tr_w1      = (const float*)d_in[P + 31];
  const float* tr_b1      = (const float*)d_in[P + 32];
  const float* tr_w2      = (const float*)d_in[P + 33];
  const float* tr_b2      = (const float*)d_in[P + 34];
  const float* tr_fc_w    = (const float*)d_in[P + 35];
  const float* tr_fc_b    = (const float*)d_in[P + 36];
  const float* st_wih     = (const float*)d_in[P + 37];
  const float* st_whh     = (const float*)d_in[P + 38];
  const float* st_bih     = (const float*)d_in[P + 39];
  const float* st_bhh     = (const float*)d_in[P + 40];

  float* out = (float*)d_out;
  const size_t o_z  = 0;
  const size_t o_q  = 320;
  const size_t o_lq = 640;
  const size_t o_p  = 960;
  const size_t o_lp = 1280;
  const size_t o_d1 = 1600;
  const size_t o_d2 = o_d1 + (size_t)Bq * Tm1 * Vv;   // 37,441,600
  const size_t o_b1 = o_d2 + (size_t)Bq * Tm1 * Vv;   // 74,881,600
  const size_t o_b2 = o_b1 + (size_t)Bq * Vv;         // 75,841,600
  const size_t o_nh = o_b2 + (size_t)Bq * Vv;         // 76,801,600
  const size_t o_nc = o_nh + (size_t)Bq * NST;        // 76,801,920

  // --- workspace carve (256B aligned) ---
  char* ws = (char*)d_ws;
  size_t off = 0;
  auto carve = [&](size_t bytes) -> char* {
    char* p = ws + off;
    off += (bytes + 255) & ~(size_t)255;
    return p;
  };
  float* cat1    = (float*)carve((size_t)Bq * 610 * 4);
  float* net1a   = (float*)carve((size_t)Bq * 400 * 4);
  float* net1    = (float*)carve((size_t)Bq * 200 * 4);
  float* logits  = (float*)carve((size_t)Bq * NST * 4);
  float* z_ws    = (float*)carve((size_t)Bq * NST * 4);
  float* net2a   = (float*)carve((size_t)Bq * 200 * 4);
  float* net2    = (float*)carve((size_t)Bq * 200 * 4);
  float* decin1  = (float*)carve((size_t)Bq * Hh * 4);
  float* net3a   = (float*)carve((size_t)Bq * 100 * 4);
  float* net3    = (float*)carve((size_t)Bq * 100 * 4);
  float* trlog   = (float*)carve((size_t)Bq * NST * 4);
  float* Xsum    = (float*)carve((size_t)Bq * NUq * 2 * Hh * 4);
  float* XpXc    = (float*)carve((size_t)Bq * NUq * 4 * 4);
  _Float16* Wcat1 = (_Float16*)carve((size_t)NG * KX * 2);
  _Float16* Wcat2 = (_Float16*)carve((size_t)NG * KX * 2);
  _Float16* A1    = (_Float16*)carve((size_t)Bq * Tm1 * KPAD_H * 2);
  _Float16* A2    = (_Float16*)carve((size_t)Bq * Tm1 * KPAD_H * 2);
  float* h1_ws   = (float*)carve((size_t)Bq * Hh * 4);
  float* dec2h   = (float*)carve((size_t)Bq * 420 * 4);
  float* bowh1   = (float*)carve((size_t)Bq * 400 * 4);
  float* bowh2   = (float*)carve((size_t)Bq * 400 * 4);
  _Float16* bowA1 = (_Float16*)carve((size_t)Bq * KPAD_B * 2);
  _Float16* bowA2 = (_Float16*)carve((size_t)Bq * KPAD_B * 2);
  _Float16* fc1t  = (_Float16*)carve((size_t)Vv * KPAD_H * 2);
  _Float16* fc2t  = (_Float16*)carve((size_t)Vv * KPAD_H * 2);
  _Float16* bp1t  = (_Float16*)carve((size_t)Vv * KPAD_B * 2);
  _Float16* bp2t  = (_Float16*)carve((size_t)Vv * KPAD_B * 2);

  const unsigned TB = 256;

  // ===== Encoder =====
  k_concat2<<<cdiv((size_t)Bq * 610, TB), TB, 0, stream>>>(state_h, NST, inputs, 600, cat1, Bq);
  k_sgemm_act<<<cdiv((size_t)Bq * 400, TB), TB, 0, stream>>>(cat1, enc_w1, enc_b1, net1a, Bq, 610, 400, 1);
  k_sgemm_act<<<cdiv((size_t)Bq * 200, TB), TB, 0, stream>>>(net1a, enc_w2, enc_b2, net1, Bq, 400, 200, 1);
  k_sgemm_act<<<cdiv((size_t)Bq * NST, TB), TB, 0, stream>>>(net1, enc_fc_w, enc_fc_b, logits, Bq, 200, NST, 0);
  k_enc_softmax<<<1, 32, 0, stream>>>(logits, gumbel, p_tau,
                                      out + o_z, out + o_q, out + o_lq, z_ws);
  k_sgemm_act<<<cdiv((size_t)Bq * 200, TB), TB, 0, stream>>>(z_ws, dec_w1, dec_b1, net2a, Bq, NST, 200, 1);
  k_sgemm_act<<<cdiv((size_t)Bq * 200, TB), TB, 0, stream>>>(net2a, dec_w2, dec_b2, net2, Bq, 200, 200, 1);
  k_concat2<<<cdiv((size_t)Bq * Hh, TB), TB, 0, stream>>>(state_h, NST, net2, 200, decin1, Bq);

  // ===== Transition prior + state LSTM =====
  k_sgemm_act<<<cdiv((size_t)Bq * 100, TB), TB, 0, stream>>>(state_h, tr_w1, tr_b1, net3a, Bq, NST, 100, 1);
  k_sgemm_act<<<cdiv((size_t)Bq * 100, TB), TB, 0, stream>>>(net3a, tr_w2, tr_b2, net3, Bq, 100, 100, 1);
  k_sgemm_act<<<cdiv((size_t)Bq * NST, TB), TB, 0, stream>>>(net3, tr_fc_w, tr_fc_b, trlog, Bq, 100, NST, 0);
  k_pz<<<1, 32, 0, stream>>>(trlog, out + o_p, out + o_lp);
  k_state_lstm<<<cdiv((size_t)Bq * NST, TB), TB, 0, stream>>>(net2, inputs, state_h, state_c,
                                                              st_wih, st_whh, st_bih, st_bhh,
                                                              out + o_nh, out + o_nc);

  // ===== Decoder prep =====
  k_xsum<<<cdiv((size_t)Bq * NUq * 2 * Hh, TB), TB, 0, stream>>>(query, Xsum);
  k_xpxc<<<cdiv((size_t)Bq * NUq * 4, TB), TB, 0, stream>>>(query, XpXc);
  k_wcat<<<cdiv((size_t)NG * KX, TB), TB, 0, stream>>>(d1_wih, d1_whh, Wcat1);
  k_wcat<<<cdiv((size_t)NG * KX, TB), TB, 0, stream>>>(d2_wih, d2_whh, Wcat2);
  k_zero_f16<<<cdiv((size_t)Bq * Tm1 * KPAD_H, TB), TB, 0, stream>>>(A1, Bq * Tm1 * KPAD_H);
  k_zero_f16<<<cdiv((size_t)Bq * Tm1 * KPAD_H, TB), TB, 0, stream>>>(A2, Bq * Tm1 * KPAD_H);
  k_transpose_f16<<<cdiv((size_t)Vv * KPAD_H, TB), TB, 0, stream>>>(fc1_w, fc1t, Hh, Vv, KPAD_H);
  k_transpose_f16<<<cdiv((size_t)Vv * KPAD_H, TB), TB, 0, stream>>>(fc2_w, fc2t, Hh, Vv, KPAD_H);
  k_transpose_f16<<<cdiv((size_t)Vv * KPAD_B, TB), TB, 0, stream>>>(bow_p1_w, bp1t, 400, Vv, KPAD_B);
  k_transpose_f16<<<cdiv((size_t)Vv * KPAD_B, TB), TB, 0, stream>>>(bow_p2_w, bp2t, 400, Vv, KPAD_B);

  // ===== Sequential decoders (persistent WGP kernel, WMMA inside) =====
  k_decoder<<<1, 1024, DEC_LDS_BYTES, stream>>>(decin1, emb1, emb2, prevemb, Xsum, XpXc,
                                                Wcat1, Wcat2, d1_bih, d1_bhh, d2_bih, d2_bhh,
                                                A1, A2, h1_ws);

  // ===== BoW heads (first layers) =====
  k_concat2<<<cdiv((size_t)Bq * 420, TB), TB, 0, stream>>>(decin1, Hh, h1_ws, Hh, dec2h, Bq);
  k_sgemm_act<<<cdiv((size_t)Bq * 400, TB), TB, 0, stream>>>(decin1, bow_fc1_w, bow_fc1_b, bowh1, Bq, Hh, 400, 2);
  k_sgemm_act<<<cdiv((size_t)Bq * 400, TB), TB, 0, stream>>>(dec2h, bow_fc2_w, bow_fc2_b, bowh2, Bq, 420, 400, 2);
  k_rows_to_f16<<<cdiv((size_t)Bq * KPAD_B, TB), TB, 0, stream>>>(bowh1, bowA1, Bq, 400, KPAD_B);
  k_rows_to_f16<<<cdiv((size_t)Bq * KPAD_B, TB), TB, 0, stream>>>(bowh2, bowA2, Bq, 400, KPAD_B);

  // ===== Big WMMA vocab projections =====
  {
    const int Ntiles = Vv / 16;                 // 1875
    const unsigned gx = cdiv(cdiv(Ntiles, 4), 8);   // 59
    dim3 gridD(gx, (Bq * Tm1) / 16);            // (59, 78)
    k_wmma_gemm_bias<<<gridD, 256, 0, stream>>>(A1, fc1t, fc1_b, out + o_d1, Bq * Tm1, Vv, KPAD_H);
    k_wmma_gemm_bias<<<gridD, 256, 0, stream>>>(A2, fc2t, fc2_b, out + o_d2, Bq * Tm1, Vv, KPAD_H);
    dim3 gridB(gx, Bq / 16);                    // (59, 2)
    k_wmma_gemm_bias<<<gridB, 256, 0, stream>>>(bowA1, bp1t, bow_p1_b, out + o_b1, Bq, Vv, KPAD_B);
    k_wmma_gemm_bias<<<gridB, 256, 0, stream>>>(bowA2, bp2t, bow_p2_b, out + o_b2, Bq, Vv, KPAD_B);
  }
}
